// NeuralMemory_62654982914321
// MI455X (gfx1250) — matrix operand order repaired and checked by reading, not compiled, not run on hardware
//
#include <hip/hip_runtime.h>
#include <hip/hip_bf16.h>

// NeuralMemory pipeline for MI455X (gfx1250), wave32 + WMMA bf16.
//
// h1   = relu(X @ W1^T + b1)
// h    = relu(h1 @ W2^T + b2)          -> also emit h^T (free via WMMA C layout)
// res  = (h @ Wm^T + bm) - value       -> emit c*res^T, c = lr*2/(B*D)
// part[z] = partial of (c*res^T) @ h   (split-K = 16, deterministic)
// Wu   = (1-fg)*Wm + sum_z part[z]
// out  = h @ Wu^T + bm                  (fp32)

typedef __bf16 bf16;
typedef __attribute__((ext_vector_type(8)))  __bf16 v8bf;
typedef __attribute__((ext_vector_type(16))) __bf16 v16bf;
typedef __attribute__((ext_vector_type(8)))  float  v8f;

#define DMEM   512
#define BROWS  32768
#define SPLITK 16

// Build a 16-element bf16 fragment from two contiguous 16B chunks.
__device__ __forceinline__ v16bf frag16(const bf16* p, int o0, int o1) {
  v8bf a = *(const v8bf*)(p + o0);
  v8bf b = *(const v8bf*)(p + o1);
  return __builtin_shufflevector(a, b, 0,1,2,3,4,5,6,7,8,9,10,11,12,13,14,15);
}

enum { EPI_RELU = 0, EPI_RELU_BOTH = 1, EPI_RESID_T = 2, EPI_GRAD_PART = 3, EPI_OUT_F32 = 4 };

// NT GEMM: D[r, c] = sum_k A[r, k] * Bw[c, k]   (A: rows x K, Bw: 512 x K, both K-contiguous)
// Block = 256 threads = 8 waves; block tile = 128 rows x 64 cols; wave tile = 16 x 64.
template <int MODE>
__global__ __launch_bounds__(256) void gemm_nt_wmma(
    const bf16* __restrict__ A, const bf16* __restrict__ Bw,
    const float* __restrict__ bias, int K, int kChunk,
    const float* __restrict__ value, const float* __restrict__ lr,
    bf16* __restrict__ outB, bf16* __restrict__ outT, int ldT,
    float* __restrict__ outF)
{
  const int lane  = threadIdx.x & 31;
  const int wave  = threadIdx.x >> 5;
  const int laneM = lane & 15;
  const int hi    = lane >> 4;          // which half-wave (selects K sub-range)
  const int row0  = blockIdx.y * 128 + wave * 16;
  const int cb    = blockIdx.x * 64;
  const int k0    = blockIdx.z * kChunk;

  v8f acc[4] = {};

  const bf16* arow = A  + (size_t)(row0 + laneM) * (size_t)K + k0;
  const bf16* brow = Bw + (size_t)(cb   + laneM) * (size_t)K + k0;
  const size_t bstep = (size_t)16 * (size_t)K;

  for (int kk = 0; kk < kChunk; kk += 32) {
    // A 16x32 bf16 fragment: lanes 0-15 hold K[0..7],[16..23]; lanes 16-31 K[8..15],[24..31]
    v16bf af = frag16(arow + kk, hi * 8, 16 + hi * 8);
    __builtin_prefetch(arow + kk + 256, 0, 1);   // global_prefetch_b8, speculative
#pragma unroll
    for (int t = 0; t < 4; ++t) {
      // B 32x16 fragment: lane = column, lanes 0-15 hold K[0..7],[16..23]; lanes 16-31 K[8..15],[24..31]
      v16bf bw = frag16(brow + (size_t)t * bstep + kk, hi * 8, 16 + hi * 8);
      acc[t] = __builtin_amdgcn_wmma_f32_16x16x32_bf16(
          false, af, false, bw, (short)0, acc[t], false, false);
    }
  }

  // C/D layout: acc[t][j] is element (M = row0 + hi*8 + j, N = cb + t*16 + laneM).
  // Per lane the 8 values span 8 CONSECUTIVE rows -> one b128 store for transposed output.
#pragma unroll
  for (int t = 0; t < 4; ++t) {
    const int col = cb + t * 16 + laneM;
    if (MODE == EPI_RELU || MODE == EPI_RELU_BOTH) {
      const float bv = bias[col];
      v8bf pk;
#pragma unroll
      for (int j = 0; j < 8; ++j) {
        const int r = row0 + hi * 8 + j;
        float v = acc[t][j] + bv;
        v = v > 0.0f ? v : 0.0f;
        const bf16 q = (bf16)v;
        outB[(size_t)r * DMEM + col] = q;
        pk[j] = q;
      }
      if (MODE == EPI_RELU_BOTH)
        *(v8bf*)(outT + (size_t)col * (size_t)ldT + (row0 + hi * 8)) = pk;
    } else if (MODE == EPI_RESID_T) {
      const float bv = bias[col];
      const float c  = lr[0] * (2.0f / ((float)BROWS * (float)DMEM));
      v8bf pk;
#pragma unroll
      for (int j = 0; j < 8; ++j) {
        const int r = row0 + hi * 8 + j;
        const float v = acc[t][j] + bv - value[(size_t)r * DMEM + col];
        pk[j] = (bf16)(v * c);
      }
      *(v8bf*)(outT + (size_t)col * (size_t)ldT + (row0 + hi * 8)) = pk;
    } else if (MODE == EPI_GRAD_PART) {
      // Deterministic split-K: each z-slice writes its own 512x512 fp32 partial.
      float* part = outF + (size_t)blockIdx.z * (DMEM * DMEM);
#pragma unroll
      for (int j = 0; j < 8; ++j) {
        const int r = row0 + hi * 8 + j;
        part[(size_t)r * DMEM + col] = acc[t][j];
      }
    } else { // EPI_OUT_F32
      const float bv = bias[col];
#pragma unroll
      for (int j = 0; j < 8; ++j) {
        const int r = row0 + hi * 8 + j;
        outF[(size_t)r * DMEM + col] = acc[t][j] + bv;
      }
    }
  }
}

__global__ void f32_to_bf16_k(const float* __restrict__ in, bf16* __restrict__ out, int n) {
  const int i = blockIdx.x * blockDim.x + threadIdx.x;
  if (i < n) out[i] = (bf16)in[i];
}

// Wu = (1-fg)*Wm + sum over SPLITK partials  (fixed summation order -> bit-exact replays)
__global__ void wupdate_reduce_k(const float* __restrict__ memW,
                                 const float* __restrict__ part,
                                 const float* __restrict__ fg,
                                 bf16* __restrict__ out, int n) {
  const int i = blockIdx.x * blockDim.x + threadIdx.x;
  if (i < n) {
    float s = 0.0f;
#pragma unroll
    for (int z = 0; z < SPLITK; ++z) s += part[(size_t)z * n + i];
    out[i] = (bf16)((1.0f - fg[0]) * memW[i] + s);
  }
}

extern "C" void kernel_launch(void* const* d_in, const int* in_sizes, int n_in,
                              void* d_out, int out_size, void* d_ws, size_t ws_size,
                              hipStream_t stream) {
  const float* key_x = (const float*)d_in[0];
  const float* value = (const float*)d_in[1];
  const float* W1    = (const float*)d_in[2];
  const float* b1    = (const float*)d_in[3];
  const float* W2    = (const float*)d_in[4];
  const float* b2    = (const float*)d_in[5];
  const float* mem_W = (const float*)d_in[6];
  const float* mem_b = (const float*)d_in[7];
  const float* fg    = (const float*)d_in[8];
  const float* lr    = (const float*)d_in[9];
  float* out = (float*)d_out;

  const size_t SB = (size_t)BROWS * DMEM;   // 16M activation elements
  const int NX = BROWS * DMEM;
  const int NW = DMEM * DMEM;

  char* w = (char*)d_ws;
  bf16* bufX  = (bf16*)w; w += SB * 2;      // X(bf16), then reused as H
  bf16* bufH1 = (bf16*)w; w += SB * 2;      // H1, then reused as c*resid^T
  bf16* bufHT = (bf16*)w; w += SB * 2;      // H^T
  bf16* W1b   = (bf16*)w; w += (size_t)NW * 2;
  bf16* W2b   = (bf16*)w; w += (size_t)NW * 2;
  bf16* Wmb   = (bf16*)w; w += (size_t)NW * 2;
  bf16* Wub   = (bf16*)w; w += (size_t)NW * 2;
  float* gpart = (float*)w; w += (size_t)SPLITK * NW * 4;   // 16 MB of split-K partials

  // Precision staging
  f32_to_bf16_k<<<(NX + 255) / 256, 256, 0, stream>>>(key_x, bufX, NX);
  f32_to_bf16_k<<<(NW + 255) / 256, 256, 0, stream>>>(W1, W1b, NW);
  f32_to_bf16_k<<<(NW + 255) / 256, 256, 0, stream>>>(W2, W2b, NW);
  f32_to_bf16_k<<<(NW + 255) / 256, 256, 0, stream>>>(mem_W, Wmb, NW);

  dim3 blk(256);
  dim3 gBig(DMEM / 64, BROWS / 128, 1);     // (8, 256)
  // h1 = relu(X W1^T + b1)
  gemm_nt_wmma<EPI_RELU><<<gBig, blk, 0, stream>>>(
      bufX, W1b, b1, DMEM, DMEM, nullptr, nullptr, bufH1, nullptr, 0, nullptr);
  // h = relu(h1 W2^T + b2); also h^T
  gemm_nt_wmma<EPI_RELU_BOTH><<<gBig, blk, 0, stream>>>(
      bufH1, W2b, b2, DMEM, DMEM, nullptr, nullptr, bufX, bufHT, BROWS, nullptr);
  // c*resid^T, c = lr*2/(B*D)
  gemm_nt_wmma<EPI_RESID_T><<<gBig, blk, 0, stream>>>(
      bufX, Wmb, mem_b, DMEM, DMEM, value, lr, nullptr, bufH1, BROWS, nullptr);
  // grad partials: (c*resid^T) @ (h^T)^T, NT GEMM over K = 32768, split-K = 16
  dim3 gGrad(DMEM / 64, DMEM / 128, SPLITK);  // (8, 4, 16), kChunk = 2048
  gemm_nt_wmma<EPI_GRAD_PART><<<gGrad, blk, 0, stream>>>(
      bufH1, bufHT, nullptr, BROWS, BROWS / SPLITK, nullptr, nullptr, nullptr, nullptr, 0, gpart);
  // Wu = (1-fg)*Wm + sum_z part[z]
  wupdate_reduce_k<<<(NW + 255) / 256, 256, 0, stream>>>(mem_W, gpart, fg, Wub, NW);
  // out = h Wu^T + mem_b (fp32)
  gemm_nt_wmma<EPI_OUT_F32><<<gBig, blk, 0, stream>>>(
      bufX, Wub, mem_b, DMEM, DMEM, nullptr, nullptr, nullptr, nullptr, 0, out);
}
